// SensorStructureModel_7516192768685
// MI455X (gfx1250) — matrix operand order repaired and checked by reading, not compiled, not run on hardware
//
#include <hip/hip_runtime.h>
#include <hip/hip_bf16.h>

#define DEVINL __device__ __forceinline__

typedef __attribute__((ext_vector_type(16))) __bf16 v16bf;
typedef __attribute__((ext_vector_type(8)))  float  v8f;

constexpr int N = 96, T = 8192, E = 128, C = 64, L = 32;

// ---------------- workspace layout (float elements) ----------------
constexpr size_t OFF_TCOMP = 0;                               // 96*64*32
constexpr size_t OFF_X     = OFF_TCOMP + (size_t)N * C * L;   // 96*128
constexpr size_t OFF_TFEAT = OFF_X     + (size_t)N * E;       // 96*128*32
constexpr size_t OFF_Q     = OFF_TFEAT + (size_t)N * E * L;
constexpr size_t OFF_K     = OFF_Q     + (size_t)N * E;
constexpr size_t OFF_V     = OFF_K     + (size_t)N * E;
constexpr size_t OFF_H1    = OFF_V     + (size_t)N * E;
constexpr size_t OFF_H2    = OFF_H1    + (size_t)N * E;
constexpr size_t OFF_A1    = OFF_H2    + (size_t)N * E;       // 96*96 scratch
constexpr size_t OFF_EU    = OFF_A1    + (size_t)N * N;       // 96*128
constexpr size_t OFF_EV    = OFF_EU    + (size_t)N * E;
constexpr size_t OFF_OU    = OFF_EV    + (size_t)N * E;       // 96*64
constexpr size_t OFF_OV    = OFF_OU    + (size_t)N * (E / 2);
constexpr size_t OFF_P     = OFF_OV    + (size_t)N * (E / 2); // 96*128*32
constexpr size_t OFF_QM    = OFF_P     + (size_t)N * E * L;   // 96*128*32
// pre-swizzled bf16 A-fragment tensors (counted in float slots, 2 bf16/float)
constexpr size_t OFF_PAENC = OFF_QM    + (size_t)N * E * L;   // 20480 bf16
constexpr size_t OFF_PAOV2 = OFF_PAENC + 10240;               // 24576 bf16
constexpr size_t WS_FLOATS = OFF_PAOV2 + 12288;

// ---------------- output layout (float elements) ----------------
constexpr size_t OUT_NODE = 0;                           // 96*3
constexpr size_t OUT_EDGE = OUT_NODE + (size_t)N * 3;    // 96*96
constexpr size_t OUT_ORD  = OUT_EDGE + (size_t)N * N;    // 96*96*2
constexpr size_t OUT_BIT  = OUT_ORD  + (size_t)N * N * 2;// 96*16
constexpr size_t OUT_OVF  = OUT_BIT  + (size_t)N * 16;   // 96*96*32
constexpr size_t OUT_A2   = OUT_OVF  + (size_t)N * N * L;// 96*96

// ---------------- WMMA helpers ----------------
DEVINL v8f wmma_bf16(v16bf a, v16bf b, v8f c) {
  return __builtin_amdgcn_wmma_f32_16x16x32_bf16(false, a, false, b, (short)0, c, false, false);
}

// 32 contiguous bytes of bf16 (LDS or global), 16B-aligned.
DEVINL v16bf ld32B(const __bf16* p) {
  union { uint4 u[2]; v16bf v; } r;
  const uint4* q = (const uint4*)p;
  r.u[0] = q[0];
  r.u[1] = q[1];
  return r.v;
}

// Direct A fragment from a global f32 weight tensor W[co][ci][k]:
//   element(idx) -> ci = ciBase + q*32 + 16*half + idx
DEVINL v16bf ldA(const float* __restrict__ w, int rowStride, int ciStride,
                 int co, int ciBase, int k, int q, int h) {
  v16bf a;
#pragma unroll
  for (int i = 0; i < 8; ++i) {
    int ci0 = ciBase + q * 32 + 16 * h + i;
    int ci1 = ci0 + 8;
    a[i]     = (__bf16)w[co * rowStride + ci0 * ciStride + k];
    a[i + 8] = (__bf16)w[co * rowStride + ci1 * ciStride + k];
  }
  return a;
}

DEVINL float halfReduce16(float v) {  // sum across the 16 lanes of a half-wave
  v += __shfl_xor(v, 1);
  v += __shfl_xor(v, 2);
  v += __shfl_xor(v, 4);
  v += __shfl_xor(v, 8);
  return v;
}

// =====================================================================
// Pre-swizzle: W[co][ci][k] (f32) -> lane-ordered bf16 A fragments.
// Consumer loads fragment f for lane as 32 contiguous bytes at
//   out[((cotile*F + f)*32 + lane)*16]      with f = k*qchunks + q.
// =====================================================================
__global__ __launch_bounds__(256) void preswz_kernel(
    const float* __restrict__ w, __bf16* __restrict__ outp,
    int rowStride, int ciStride, int ciBase,
    int numCotiles, int taps, int qchunks) {
  const int F = taps * qchunks;
  const int total = numCotiles * F * 32 * 16;
  for (int idx = blockIdx.x * blockDim.x + threadIdx.x; idx < total;
       idx += gridDim.x * blockDim.x) {
    int i      = idx & 15;
    int lane   = (idx >> 4) & 31;
    int f      = (idx >> 9) % F;
    int cotile = idx / (F * 512);
    int k = f / qchunks, q = f % qchunks;
    int m = lane & 15, h = lane >> 4;
    int co = cotile * 16 + m;
    int ci = ciBase + q * 32 + 16 * h + i;
    outp[idx] = (__bf16)w[co * rowStride + ci * ciStride + k];
  }
}

// =====================================================================
// Kernel 1: fused encoder.  One block per (l-block of 256 t, n).
// conv1 (VALU, h1 -> LDS bf16), conv2 via WMMA, reduce to tcomp[n][c][l].
// h2 is never materialized in global memory.
// =====================================================================
__global__ __launch_bounds__(256) void enc_kernel(
    const float* __restrict__ raw, const float* __restrict__ w1,
    const float* __restrict__ b1, const __bf16* __restrict__ paEnc,
    const float* __restrict__ b2, float* __restrict__ tcomp) {
  const int l  = blockIdx.x;   // 0..31
  const int n  = blockIdx.y;   // 0..95
  const int t0 = l * 256;
  const int tid  = threadIdx.x;
  const int lane = tid & 31, wave = tid >> 5;
  const int hh = (lane >> 4), mm = lane & 15;

  __shared__ __align__(16) __bf16 h1S[260][72];  // [t_local][ci], pad stride 72
  __shared__ float rawS[272];
  __shared__ float w1S[64 * 7];
  __shared__ float b1S[64], b2S[64];
  __shared__ float accS[64];

  // stage raw chunk with zero padding: rawS[s] = raw[t0-5+s]
  for (int s = tid; s < 266; s += 256) {
    int t = t0 - 5 + s;
    rawS[s] = (t >= 0 && t < T) ? raw[n * T + t] : 0.0f;
  }
  for (int s = tid; s < 448; s += 256) w1S[s] = w1[s];
  if (tid < 64) { b1S[tid] = b1[tid]; b2S[tid] = b2[tid]; accS[tid] = 0.0f; }
  __syncthreads();

  // h1 = relu(conv1(raw)) for t in [t0-2, t0+258), as bf16 in LDS
  for (int idx = tid; idx < 260 * 64; idx += 256) {
    int c = idx & 63, tl = idx >> 6;
    float v = b1S[c];
#pragma unroll
    for (int k = 0; k < 7; ++k) v += rawS[tl + k] * w1S[c * 7 + k];
    int tg = t0 - 2 + tl;
    v = fmaxf(v, 0.0f);
    h1S[tl][c] = (tg >= 0 && tg < T) ? (__bf16)v : (__bf16)0.0f;
  }
  __syncthreads();

  // conv2 via WMMA: h2[co,t] = sum_{k,ci} W2[co,ci,k]*h1[ci,t+k-2]
  const int cotile = wave & 3;
  const __bf16* pa = paEnc + ((size_t)cotile * 10 * 32 + lane) * 16;
  v16bf aF[10];
#pragma unroll
  for (int f = 0; f < 10; ++f) aF[f] = ld32B(pa + (size_t)f * 512);

  float rsum[8];
#pragma unroll
  for (int r = 0; r < 8; ++r) rsum[r] = 0.0f;

  for (int tt = (wave >> 2); tt < 16; tt += 2) {
    // batch-load all B fragments for this tile, then chain WMMAs
    v16bf bF[10];
#pragma unroll
    for (int k = 0; k < 5; ++k)
#pragma unroll
      for (int q = 0; q < 2; ++q)
        bF[k * 2 + q] = ld32B(&h1S[tt * 16 + mm + k][q * 32 + 16 * hh]);
    v8f acc = {};
#pragma unroll
    for (int f = 0; f < 10; ++f) acc = wmma_bf16(aF[f], bF[f], acc);
    // bias + relu, accumulate per-channel partial sums in registers
#pragma unroll
    for (int r = 0; r < 8; ++r) {
      int co = cotile * 16 + r + 8 * hh;
      rsum[r] += fmaxf(acc[r] + b2S[co], 0.0f);
    }
  }
  // one cross-lane reduction per wave (not per tile)
#pragma unroll
  for (int r = 0; r < 8; ++r) {
    int co = cotile * 16 + r + 8 * hh;
    float v = halfReduce16(rsum[r]);
    if (mm == 0) atomicAdd(&accS[co], v);
  }
  __syncthreads();
  if (tid < 64) tcomp[(n * 64 + tid) * 32 + l] = accS[tid] * (1.0f / 256.0f);
}

// =====================================================================
// Kernel 2: X = mean_T(h2) @ fc_w + fc_b  (mean_T = mean over tcomp blocks)
// =====================================================================
__global__ __launch_bounds__(128) void xproj_kernel(
    const float* __restrict__ tcomp, const float* __restrict__ fcw,
    const float* __restrict__ fcb, float* __restrict__ Xo) {
  int n = blockIdx.x, tid = threadIdx.x;
  __shared__ float meanS[64];
  if (tid < 64) {
    float s = 0.0f;
    for (int l = 0; l < 32; ++l) s += tcomp[(n * 64 + tid) * 32 + l];
    meanS[tid] = s * (1.0f / 32.0f);
  }
  __syncthreads();
  float a = fcb[tid];
  for (int c = 0; c < 64; ++c) a += meanS[c] * fcw[c * 128 + tid];
  Xo[n * 128 + tid] = a;
}

// =====================================================================
// Kernel 3: tfeat[n,e,l] = sum_c tcomp[n,c,l]*tp_w[e,c] + tp_b[e]
// =====================================================================
__global__ __launch_bounds__(128) void tfeat_kernel(
    const float* __restrict__ tcomp, const float* __restrict__ tpw,
    const float* __restrict__ tpb, float* __restrict__ tf) {
  int n = blockIdx.x, tid = threadIdx.x;
  __shared__ float tcS[64 * 32];
  for (int idx = tid; idx < 2048; idx += 128) tcS[idx] = tcomp[n * 2048 + idx];
  __syncthreads();
  float b = tpb[tid];
  for (int l = 0; l < 32; ++l) {
    float a = b;
    for (int c = 0; c < 64; ++c) a += tcS[c * 32 + l] * tpw[tid * 64 + c];
    tf[n * 4096 + tid * 32 + l] = a;
  }
}

// =====================================================================
// Graph layer: Q/K/V projections (fp32 VALU, tiny)
// =====================================================================
__global__ __launch_bounds__(128) void qkv_kernel(
    const float* __restrict__ In,
    const float* __restrict__ qw, const float* __restrict__ qb,
    const float* __restrict__ kw, const float* __restrict__ kb,
    const float* __restrict__ vw, const float* __restrict__ vb,
    float* __restrict__ Qo, float* __restrict__ Ko, float* __restrict__ Vo) {
  int n = blockIdx.x, tid = threadIdx.x;
  __shared__ float xr[128];
  xr[tid] = In[n * 128 + tid];
  __syncthreads();
  float q = qb[tid], k = kb[tid], v = vb[tid];
  for (int f = 0; f < 128; ++f) {
    float x = xr[f];
    q += x * qw[f * 128 + tid];
    k += x * kw[f * 128 + tid];
    v += x * vw[f * 128 + tid];
  }
  Qo[n * 128 + tid] = q;
  Ko[n * 128 + tid] = k;
  Vo[n * 128 + tid] = v;
}

// =====================================================================
// Graph layer: attention row + output proj. One block per row i.
// =====================================================================
__global__ __launch_bounds__(128) void attn_kernel(
    const float* __restrict__ Q, const float* __restrict__ K,
    const float* __restrict__ V, const float* __restrict__ Xin,
    const float* __restrict__ ow, const float* __restrict__ ob,
    float* __restrict__ Aout, float* __restrict__ Hout) {
  int i = blockIdx.x, tid = threadIdx.x;
  __shared__ float qrow[128], sc[96], mrow[128];
  __shared__ float smax, sinv;
  qrow[tid] = Q[i * 128 + tid];
  __syncthreads();
  if (tid < 96) {
    float s = 0.0f;
    for (int e = 0; e < 128; ++e) s += qrow[e] * K[tid * 128 + e];
    sc[tid] = s * 0.08838834764831845f;  // 1/sqrt(128)
  }
  __syncthreads();
  if (tid == 0) {
    float m = sc[0];
    for (int j = 1; j < 96; ++j) m = fmaxf(m, sc[j]);
    smax = m;
  }
  __syncthreads();
  if (tid < 96) sc[tid] = expf(sc[tid] - smax);
  __syncthreads();
  if (tid == 0) {
    float s = 0.0f;
    for (int j = 0; j < 96; ++j) s += sc[j];
    sinv = 1.0f / s;
  }
  __syncthreads();
  if (tid < 96) { sc[tid] *= sinv; Aout[i * 96 + tid] = sc[tid]; }
  __syncthreads();
  float mv = Xin[i * 128 + tid];
  for (int j = 0; j < 96; ++j) mv += sc[j] * V[j * 128 + tid];
  mrow[tid] = mv;
  __syncthreads();
  float o = ob[tid];
  for (int f = 0; f < 128; ++f) o += mrow[f] * ow[f * 128 + tid];
  Hout[i * 128 + tid] = fmaxf(o, 0.0f);
}

// =====================================================================
// node_logits = H2 @ node_w + node_b
// =====================================================================
__global__ __launch_bounds__(288) void node_kernel(
    const float* __restrict__ H2, const float* __restrict__ nw,
    const float* __restrict__ nb, float* __restrict__ out) {
  int idx = threadIdx.x;
  if (idx >= 288) return;
  int n = idx / 3, o = idx % 3;
  float a = nb[o];
  for (int f = 0; f < 128; ++f) a += H2[n * 128 + f] * nw[f * 3 + o];
  out[n * 3 + o] = a;
}

// =====================================================================
// bitmask_logits = relu(H2@W1+b1)@W2+b2
// =====================================================================
__global__ __launch_bounds__(64) void bitmask_kernel(
    const float* __restrict__ H2, const float* __restrict__ w1,
    const float* __restrict__ b1, const float* __restrict__ w2,
    const float* __restrict__ b2, float* __restrict__ out) {
  int n = blockIdx.x, tid = threadIdx.x;
  __shared__ float hb[64];
  float a = b1[tid];
  for (int f = 0; f < 128; ++f) a += H2[n * 128 + f] * w1[f * 64 + tid];
  hb[tid] = fmaxf(a, 0.0f);
  __syncthreads();
  if (tid < 16) {
    float o = b2[tid];
    for (int c = 0; c < 64; ++c) o += hb[c] * w2[c * 16 + tid];
    out[n * 16 + tid] = o;
  }
}

// =====================================================================
// Edge factorization: U = H2@W1[:128], Vv = H2@W1[128:]
// =====================================================================
__global__ __launch_bounds__(128) void edgeuv_kernel(
    const float* __restrict__ H2, const float* __restrict__ w1,
    float* __restrict__ U, float* __restrict__ Vv) {
  int n = blockIdx.x, tid = threadIdx.x;
  __shared__ float hr[128];
  hr[tid] = H2[n * 128 + tid];
  __syncthreads();
  float u = 0.0f, v = 0.0f;
  for (int f = 0; f < 128; ++f) {
    float x = hr[f];
    u += x * w1[f * 128 + tid];
    v += x * w1[(128 + f) * 128 + tid];
  }
  U[n * 128 + tid] = u;
  Vv[n * 128 + tid] = v;
}

__global__ __launch_bounds__(64) void orduv_kernel(
    const float* __restrict__ H2, const float* __restrict__ w1,
    float* __restrict__ U, float* __restrict__ Vv) {
  int n = blockIdx.x, tid = threadIdx.x;
  __shared__ float hr[128];
  hr[tid] = H2[n * 128 + tid];
  hr[64 + tid] = H2[n * 128 + 64 + tid];
  __syncthreads();
  float u = 0.0f, v = 0.0f;
  for (int f = 0; f < 128; ++f) {
    float x = hr[f];
    u += x * w1[f * 64 + tid];
    v += x * w1[(128 + f) * 64 + tid];
  }
  U[n * 64 + tid] = u;
  Vv[n * 64 + tid] = v;
}

// =====================================================================
// Per-pair edge + order logits from factorized projections.
// =====================================================================
__global__ __launch_bounds__(128) void pair_kernel(
    const float* __restrict__ U, const float* __restrict__ Vv,
    const float* __restrict__ eb1, const float* __restrict__ ew2,
    const float* __restrict__ eb2,
    const float* __restrict__ Uo, const float* __restrict__ Vo,
    const float* __restrict__ ob1, const float* __restrict__ ow2,
    const float* __restrict__ ob2,
    float* __restrict__ edge_out, float* __restrict__ ord_out) {
  int j = blockIdx.x, i = blockIdx.y, tid = threadIdx.x;
  __shared__ float red[128], red2[128];
  float he = fmaxf(U[i * 128 + tid] + Vv[j * 128 + tid] + eb1[tid], 0.0f);
  red[tid] = he * ew2[tid];
  if (tid < 64) {
    float ho = fmaxf(Uo[i * 64 + tid] + Vo[j * 64 + tid] + ob1[tid], 0.0f);
    red2[tid]      = ho * ow2[tid * 2 + 0];
    red2[64 + tid] = ho * ow2[tid * 2 + 1];
  }
  __syncthreads();
  for (int s = 64; s > 0; s >>= 1) {
    if (tid < s) red[tid] += red[tid + s];
    __syncthreads();
  }
  for (int s = 32; s > 0; s >>= 1) {
    if (tid < s) red2[tid] += red2[tid + s];
    if (tid >= 64 && tid < 64 + s) red2[tid] += red2[tid + s];
    __syncthreads();
  }
  if (tid == 0) {
    edge_out[i * 96 + j] = red[0] + eb2[0];
    ord_out[(i * 96 + j) * 2 + 0] = red2[0] + ob2[0];
    ord_out[(i * 96 + j) * 2 + 1] = red2[64] + ob2[1];
  }
}

// =====================================================================
// Overflow stage 1 (factorized): per-node conv of ov1 halves.
// P[n]  = conv(tfeat[n], ov1_w[:, :128, :])   (hsel=0)
// Qm[n] = conv(tfeat[n], ov1_w[:, 128:, :])   (hsel=1)
// =====================================================================
__global__ __launch_bounds__(256) void pq_kernel(
    const float* __restrict__ tfeat, const float* __restrict__ ov1w,
    float* __restrict__ P, float* __restrict__ Qm) {
  const int hsel = blockIdx.x;  // 0 or 1
  const int n    = blockIdx.y;
  const int tid  = threadIdx.x;
  const int lane = tid & 31, wave = tid >> 5;
  const int hh = (lane >> 4), mm = lane & 15;

  __shared__ __align__(16) __bf16 tfS[34][136];  // [l+1][ci], zero rows 0,33

  for (int idx = tid; idx < 4096; idx += 256) {
    int ci = idx >> 5, ll = idx & 31;
    tfS[ll + 1][ci] = (__bf16)tfeat[n * 4096 + ci * 32 + ll];
  }
  for (int idx = tid; idx < 272; idx += 256) {
    int row = (idx < 136) ? 0 : 33;
    tfS[row][idx % 136] = (__bf16)0.0f;
  }
  __syncthreads();

  const int cotile = wave;             // 0..7 -> co tile
  const int co_m   = cotile * 16 + mm;
  v16bf aF[12];
#pragma unroll
  for (int k = 0; k < 3; ++k)
#pragma unroll
    for (int q = 0; q < 4; ++q)
      aF[k * 4 + q] = ldA(ov1w, 768, 3, co_m, hsel * 128, k, q, hh);

  float* __restrict__ dst = hsel ? Qm : P;
#pragma unroll
  for (int lt = 0; lt < 2; ++lt) {
    v16bf bF[12];
#pragma unroll
    for (int k = 0; k < 3; ++k)
#pragma unroll
      for (int q = 0; q < 4; ++q)
        bF[k * 4 + q] = ld32B(&tfS[lt * 16 + mm + k][q * 32 + 16 * hh]);
    v8f acc = {};
#pragma unroll
    for (int f = 0; f < 12; ++f) acc = wmma_bf16(aF[f], bF[f], acc);
#pragma unroll
    for (int r = 0; r < 8; ++r) {
      int co = cotile * 16 + r + 8 * hh;
      dst[n * 4096 + co * 32 + lt * 16 + mm] = acc[r];  // no bias here
    }
  }
}

// =====================================================================
// Overflow stage 2: per pair (i,j).
// hh1 = relu(P[i]+Qm[j]+b1) -> LDS bf16; hh2 = relu(conv(hh1, ov2)+b2)
// via WMMA (pre-swizzled A); out[l] = sum_c hh2[c,l]*w3[c] + b3.
// =====================================================================
__global__ __launch_bounds__(256) void ovf_kernel(
    const float* __restrict__ P, const float* __restrict__ Qm,
    const float* __restrict__ ov1b, const __bf16* __restrict__ paOv2,
    const float* __restrict__ ov2b, const float* __restrict__ ov3w,
    const float* __restrict__ ov3b, float* __restrict__ out) {
  const int j = blockIdx.x, i = blockIdx.y;
  const int tid  = threadIdx.x;
  const int lane = tid & 31, wave = tid >> 5;
  const int hh = (lane >> 4), mm = lane & 15;

  __shared__ __align__(16) __bf16 hS[34][136];
  __shared__ float outAcc[32];
  __shared__ float b2S[64], w3S[64];

  for (int idx = tid; idx < 4096; idx += 256) {
    int ci = idx >> 5, ll = idx & 31;
    float v = P[i * 4096 + ci * 32 + ll] + Qm[j * 4096 + ci * 32 + ll] + ov1b[ci];
    hS[ll + 1][ci] = (__bf16)fmaxf(v, 0.0f);
  }
  for (int idx = tid; idx < 272; idx += 256) {
    int row = (idx < 136) ? 0 : 33;
    hS[row][idx % 136] = (__bf16)0.0f;
  }
  if (tid < 32) outAcc[tid] = 0.0f;
  if (tid >= 64 && tid < 128) { b2S[tid - 64] = ov2b[tid - 64]; w3S[tid - 64] = ov3w[tid - 64]; }
  __syncthreads();

  const int c2t = wave >> 1;   // co2 tile 0..3
  const int lt  = wave & 1;    // l tile 0..1
  const __bf16* pa = paOv2 + ((size_t)c2t * 12 * 32 + lane) * 16;
  v16bf aF[12];
#pragma unroll
  for (int f = 0; f < 12; ++f) aF[f] = ld32B(pa + (size_t)f * 512);

  v16bf bF[12];
#pragma unroll
  for (int k = 0; k < 3; ++k)
#pragma unroll
    for (int q = 0; q < 4; ++q)
      bF[k * 4 + q] = ld32B(&hS[lt * 16 + mm + k][q * 32 + 16 * hh]);
  v8f acc = {};
#pragma unroll
  for (int f = 0; f < 12; ++f) acc = wmma_bf16(aF[f], bF[f], acc);

  float part = 0.0f;
#pragma unroll
  for (int r = 0; r < 8; ++r) {
    int co = c2t * 16 + r + 8 * hh;
    part += fmaxf(acc[r] + b2S[co], 0.0f) * w3S[co];
  }
  atomicAdd(&outAcc[lt * 16 + mm], part);
  __syncthreads();
  if (tid < 32) out[(i * 96 + j) * 32 + tid] = outAcc[tid] + ov3b[0];
}

// =====================================================================
// Host launcher
// =====================================================================
extern "C" void kernel_launch(void* const* d_in, const int* in_sizes, int n_in,
                              void* d_out, int out_size, void* d_ws, size_t ws_size,
                              hipStream_t stream) {
  (void)in_sizes; (void)n_in; (void)out_size; (void)ws_size;
  const float* raw    = (const float*)d_in[0];
  const float* c1w    = (const float*)d_in[1];
  const float* c1b    = (const float*)d_in[2];
  const float* c2w    = (const float*)d_in[3];
  const float* c2b    = (const float*)d_in[4];
  const float* fcw    = (const float*)d_in[5];
  const float* fcb    = (const float*)d_in[6];
  const float* tpw    = (const float*)d_in[7];
  const float* tpb    = (const float*)d_in[8];
  const float* g1qw   = (const float*)d_in[9];
  const float* g1qb   = (const float*)d_in[10];
  const float* g1kw   = (const float*)d_in[11];
  const float* g1kb   = (const float*)d_in[12];
  const float* g1vw   = (const float*)d_in[13];
  const float* g1vb   = (const float*)d_in[14];
  const float* g1ow   = (const float*)d_in[15];
  const float* g1ob   = (const float*)d_in[16];
  const float* g2qw   = (const float*)d_in[17];
  const float* g2qb   = (const float*)d_in[18];
  const float* g2kw   = (const float*)d_in[19];
  const float* g2kb   = (const float*)d_in[20];
  const float* g2vw   = (const float*)d_in[21];
  const float* g2vb   = (const float*)d_in[22];
  const float* g2ow   = (const float*)d_in[23];
  const float* g2ob   = (const float*)d_in[24];
  const float* nodew  = (const float*)d_in[25];
  const float* nodeb  = (const float*)d_in[26];
  const float* ew1    = (const float*)d_in[27];
  const float* eb1    = (const float*)d_in[28];
  const float* ew2    = (const float*)d_in[29];
  const float* eb2    = (const float*)d_in[30];
  const float* ow1    = (const float*)d_in[31];
  const float* ob1    = (const float*)d_in[32];
  const float* ow2    = (const float*)d_in[33];
  const float* ob2    = (const float*)d_in[34];
  const float* bw1    = (const float*)d_in[35];
  const float* bb1    = (const float*)d_in[36];
  const float* bw2    = (const float*)d_in[37];
  const float* bb2    = (const float*)d_in[38];
  const float* ov1w   = (const float*)d_in[39];
  const float* ov1b   = (const float*)d_in[40];
  const float* ov2w   = (const float*)d_in[41];
  const float* ov2b   = (const float*)d_in[42];
  const float* ov3w   = (const float*)d_in[43];
  const float* ov3b   = (const float*)d_in[44];

  float* ws  = (float*)d_ws;
  float* out = (float*)d_out;
  __bf16* paEnc = (__bf16*)(ws + OFF_PAENC);
  __bf16* paOv2 = (__bf16*)(ws + OFF_PAOV2);

  // 0) pre-swizzle weight fragments for the high-traffic WMMA kernels
  preswz_kernel<<<80, 256, 0, stream>>>(c2w, paEnc, 320, 5, 0, 4, 5, 2);
  preswz_kernel<<<96, 256, 0, stream>>>(ov2w, paOv2, 384, 3, 0, 4, 3, 4);

  // 1) fused encoder -> tcomp
  enc_kernel<<<dim3(32, 96), 256, 0, stream>>>(raw, c1w, c1b, paEnc, c2b,
                                               ws + OFF_TCOMP);
  // 2) X projection, 3) tfeat
  xproj_kernel<<<96, 128, 0, stream>>>(ws + OFF_TCOMP, fcw, fcb, ws + OFF_X);
  tfeat_kernel<<<96, 128, 0, stream>>>(ws + OFF_TCOMP, tpw, tpb, ws + OFF_TFEAT);

  // 4) graph layer 1
  qkv_kernel<<<96, 128, 0, stream>>>(ws + OFF_X, g1qw, g1qb, g1kw, g1kb, g1vw,
                                     g1vb, ws + OFF_Q, ws + OFF_K, ws + OFF_V);
  attn_kernel<<<96, 128, 0, stream>>>(ws + OFF_Q, ws + OFF_K, ws + OFF_V,
                                      ws + OFF_X, g1ow, g1ob, ws + OFF_A1,
                                      ws + OFF_H1);
  // 5) graph layer 2 (A2 goes straight to output)
  qkv_kernel<<<96, 128, 0, stream>>>(ws + OFF_H1, g2qw, g2qb, g2kw, g2kb, g2vw,
                                     g2vb, ws + OFF_Q, ws + OFF_K, ws + OFF_V);
  attn_kernel<<<96, 128, 0, stream>>>(ws + OFF_Q, ws + OFF_K, ws + OFF_V,
                                      ws + OFF_H1, g2ow, g2ob, out + OUT_A2,
                                      ws + OFF_H2);

  // 6) heads
  node_kernel<<<1, 288, 0, stream>>>(ws + OFF_H2, nodew, nodeb, out + OUT_NODE);
  bitmask_kernel<<<96, 64, 0, stream>>>(ws + OFF_H2, bw1, bb1, bw2, bb2,
                                        out + OUT_BIT);
  edgeuv_kernel<<<96, 128, 0, stream>>>(ws + OFF_H2, ew1, ws + OFF_EU,
                                        ws + OFF_EV);
  orduv_kernel<<<96, 64, 0, stream>>>(ws + OFF_H2, ow1, ws + OFF_OU,
                                      ws + OFF_OV);
  pair_kernel<<<dim3(96, 96), 128, 0, stream>>>(
      ws + OFF_EU, ws + OFF_EV, eb1, ew2, eb2, ws + OFF_OU, ws + OFF_OV, ob1,
      ow2, ob2, out + OUT_EDGE, out + OUT_ORD);

  // 7) overflow branch: factorized conv1, then per-pair WMMA conv2 + head
  pq_kernel<<<dim3(2, 96), 256, 0, stream>>>(ws + OFF_TFEAT, ov1w, ws + OFF_P,
                                             ws + OFF_QM);
  ovf_kernel<<<dim3(96, 96), 256, 0, stream>>>(ws + OFF_P, ws + OFF_QM, ov1b,
                                               paOv2, ov2b, ov3w, ov3b,
                                               out + OUT_OVF);
}
